// MotLoss_70952859729923
// MI455X (gfx1250) — matrix operand order repaired and checked by reading, not compiled, not run on hardware
//
#include <hip/hip_runtime.h>
#include <hip/hip_bf16.h>
#include <math.h>

typedef __attribute__((ext_vector_type(16))) _Float16 v16h;
typedef __attribute__((ext_vector_type(8)))  _Float16 v8h;
typedef __attribute__((ext_vector_type(8)))  float    v8f;

#define B_     8
#define HM_C_  2
#define H_     152
#define W_     272
#define HW_    (H_*W_)
#define EMB_   128
#define KOBJ_  128
#define KG_    8192
#define NID0_  500
#define NID1_  300
#define NPAD_  512

// ---- workspace layout (bytes) ----
// acc floats: [0]=pos_loss [1]=neg_loss [2]=num_pos [3]=wh_num [4]=off_num
//             [5]=mask_sum [6]=nll0 [7]=nll1 ; ints at [8],[9] = cnt0,cnt1
#define OFF_ACC  0u
#define OFF_IDX  64u                       // 2*KG ints      = 65536
#define OFF_TGT  (OFF_IDX + 2u*KG_*4u)     // 2*KG ints      = 65536
#define OFF_W0H  (OFF_TGT + 2u*KG_*4u)     // 512*128 f16    = 131072
#define OFF_W1H  (OFF_W0H + (unsigned)(NPAD_*EMB_*2))
#define OFF_BP   (OFF_W1H + (unsigned)(NPAD_*EMB_*2))   // 2*512 f32 = 4096
#define OFF_EMB  (OFF_BP  + 2u*NPAD_*4u)   // 2*KG*128 f16   = 4 MB

// -------- init workspace: zero accumulators, tgt=-1, zero emb pads --------
__global__ void k_init(float* acc, int* tgt, _Float16* emb) {
  size_t i = (size_t)blockIdx.x * blockDim.x + threadIdx.x;
  size_t stride = (size_t)gridDim.x * blockDim.x;
  size_t n = (size_t)2 * KG_ * EMB_;
  for (size_t j = i; j < n; j += stride) emb[j] = (_Float16)0.f;
  if (i < 2u * KG_) tgt[i] = -1;
  if (i < 16) acc[i] = 0.f;
}

// -------- convert classifier weights to f16 (padded) + pad biases --------
__global__ void k_convw(const float* W0, const float* b0,
                        const float* W1, const float* b1,
                        _Float16* W0h, _Float16* W1h, float* biasp) {
  int i = blockIdx.x * 256 + threadIdx.x;
  if (i < 2 * NPAD_ * EMB_) {
    int cls = i / (NPAD_ * EMB_);
    int r   = i % (NPAD_ * EMB_);
    int j   = r / EMB_, k = r % EMB_;
    if (cls == 0) W0h[r] = (j < NID0_) ? (_Float16)W0[j * EMB_ + k] : (_Float16)0.f;
    else          W1h[r] = (j < NID1_) ? (_Float16)W1[j * EMB_ + k] : (_Float16)0.f;
  }
  if (i < 2 * NPAD_) {
    int cls = i / NPAD_, j = i % NPAD_;
    if (cls == 0) biasp[j]         = (j < NID0_) ? b0[j] : -1e30f;
    else          biasp[NPAD_ + j] = (j < NID1_) ? b1[j] : -1e30f;
  }
}

// -------- focal loss partial reduction --------
__global__ void k_focal(const float* __restrict__ hm_pred,
                        const float* __restrict__ hm, float* acc) {
  __shared__ float s0[256], s1[256], s2[256];
  int tid = threadIdx.x;
  int i = blockIdx.x * 256 + tid;
  float pl = 0.f, nl = 0.f, np = 0.f;
  if (i < B_ * HM_C_ * HW_) {
    float x = hm_pred[i];
    float p = 1.f / (1.f + expf(-x));
    p = fminf(fmaxf(p, 1e-4f), 1.f - 1e-4f);
    float g = hm[i];
    if (g == 1.f) {
      float q = 1.f - p;
      pl = logf(p) * q * q;
      np = 1.f;
    } else {
      float w = 1.f - g; w = w * w; w = w * w;
      nl = logf(1.f - p) * p * p * w;
    }
  }
  s0[tid] = pl; s1[tid] = nl; s2[tid] = np;
  __syncthreads();
  for (int s = 128; s; s >>= 1) {
    if (tid < s) { s0[tid] += s0[tid + s]; s1[tid] += s1[tid + s]; s2[tid] += s2[tid + s]; }
    __syncthreads();
  }
  if (tid == 0) {
    atomicAdd(&acc[0], s0[0]);
    atomicAdd(&acc[1], s1[0]);
    atomicAdd(&acc[2], s2[0]);
  }
}

// -------- wh / off L1 losses (gathered) : one 1024-thread block --------
__global__ void k_regl1(const float* __restrict__ wh_pred,
                        const float* __restrict__ reg_pred,
                        const float* __restrict__ wh,
                        const float* __restrict__ reg,
                        const float* __restrict__ reg_mask,
                        const int*   __restrict__ ind, float* acc) {
  __shared__ float sa[1024], sb[1024], sc[1024];
  int t = threadIdx.x;                 // over B*KOBJ = 1024
  int b = t / KOBJ_;
  int id = ind[t];
  float m = reg_mask[t];
  float whn = 0.f, offn = 0.f;
  for (int c = 0; c < 2; ++c) {
    float pw = wh_pred [(size_t)b * 2 * HW_ + (size_t)c * HW_ + id];
    float pr = reg_pred[(size_t)b * 2 * HW_ + (size_t)c * HW_ + id];
    float tw = wh [(size_t)t * 2 + c];
    float tr = reg[(size_t)t * 2 + c];
    whn  += fabsf(pw * m - tw * m);
    offn += fabsf(pr * m - tr * m);
  }
  sa[t] = whn; sb[t] = offn; sc[t] = m;
  __syncthreads();
  for (int s = 512; s; s >>= 1) {
    if (t < s) { sa[t] += sa[t + s]; sb[t] += sb[t + s]; sc[t] += sc[t + s]; }
    __syncthreads();
  }
  if (t == 0) {
    atomicAdd(&acc[3], sa[0]);
    atomicAdd(&acc[4], sb[0]);
    atomicAdd(&acc[5], sc[0]);
  }
}

// -------- unordered compaction of class-masked positions --------
__global__ void k_compact(const int* __restrict__ cls_id_map,
                          const int* __restrict__ cls_tr_ids,
                          int* cnt, int* idxbuf, int* tgtbuf) {
  int i = blockIdx.x * 256 + threadIdx.x;
  if (i >= B_ * HW_) return;
  int c = cls_id_map[i];
  if (c != 0 && c != 1) return;
  int p = atomicAdd(&cnt[c], 1);
  if (p < KG_) {
    idxbuf[c * KG_ + p] = i;
    int b = i / HW_, hw = i % HW_;
    tgtbuf[c * KG_ + p] = cls_tr_ids[(size_t)b * 2 * HW_ + (size_t)c * HW_ + hw];
  }
}

// -------- gather features + L2 normalize + scale -> f16 emb rows --------
__global__ void k_gathernorm(const float* __restrict__ id_pred,
                             const int* __restrict__ cnt,
                             const int* __restrict__ idxbuf, _Float16* emb) {
  int cls = blockIdx.y;
  int row = blockIdx.x;
  int n = min(cnt[cls], KG_);
  if (row >= n) return;
  int e = threadIdx.x;                 // 128 threads = EMB
  int flat = idxbuf[cls * KG_ + row];
  int b = flat / HW_, hw = flat % HW_;
  float v = id_pred[(size_t)b * EMB_ * HW_ + (size_t)e * HW_ + hw];
  __shared__ float red[128];
  red[e] = v * v;
  __syncthreads();
  for (int s = 64; s; s >>= 1) {
    if (e < s) red[e] += red[e + s];
    __syncthreads();
  }
  float norm = sqrtf(red[0]);
  float nid = (cls == 0) ? (float)NID0_ : (float)NID1_;
  float scale = sqrtf(2.f) * logf(nid - 1.f) / fmaxf(norm, 1e-12f);
  emb[(size_t)cls * KG_ * EMB_ + (size_t)row * EMB_ + e] = (_Float16)(v * scale);
}

// -------- fused WMMA GEMM (16 rows x 512 cols, K=128) + log-softmax + NLL ----
__global__ __launch_bounds__(32)
void k_gemm(const _Float16* __restrict__ emb, const _Float16* __restrict__ Wh,
            const float* __restrict__ biasp, const int* __restrict__ tgt,
            const int* __restrict__ cnt_ptr, float* nll_acc) {
  int cnt = min(*cnt_ptr, KG_);
  int rowBase = blockIdx.x * 16;
  if (rowBase >= cnt) return;

  int lane = threadIdx.x;
  int m  = lane & 15;        // A: row M ; B: col N ; D: col N
  int hi = lane >> 4;

  __shared__ float lds[16][NPAD_];

  // Preload A fragments for all 4 K-steps (K = 4*32 = 128).
  // A layout (16x32 f16): K = e + 16*hi
  const _Float16* arow = emb + (size_t)(rowBase + m) * EMB_;
  v16h afrag[4];
#pragma unroll
  for (int ks = 0; ks < 4; ++ks) {
    v8h lo = *(const v8h*)(arow + ks * 32 + hi * 16);
    v8h hh = *(const v8h*)(arow + ks * 32 + hi * 16 + 8);
    v16h a;
#pragma unroll
    for (int q = 0; q < 8; ++q) { a[q] = lo[q]; a[8 + q] = hh[q]; }
    afrag[ks] = a;
  }

  // 32 column tiles of 16 -> 512 padded columns
  for (int t = 0; t < NPAD_ / 16; ++t) {
    int col = t * 16 + m;                         // B: N = lane&15
    const _Float16* wrow = Wh + (size_t)col * EMB_ + hi * 16;  // K = e + 16*hi
    v8f accv = {};
#pragma unroll
    for (int ks = 0; ks < 4; ++ks) {
      v16h bfrag = *(const v16h*)(wrow + ks * 32);
      accv = __builtin_amdgcn_wmma_f32_16x16x32_f16(
          /*neg_a=*/false, afrag[ks], /*neg_b=*/false, bfrag,
          /*c_mod=*/(short)0, accv, /*reuse_a=*/false, /*reuse_b=*/false);
    }
#pragma unroll
    for (int r = 0; r < 8; ++r)                   // D: M = r + 8*hi
      lds[r + 8 * hi][t * 16 + m] = accv[r];
  }
  __syncthreads();

  // per-row log-softmax + NLL (32 lanes cooperate per row)
  for (int r = 0; r < 16; ++r) {
    float mx = -3.0e38f;
    for (int j = lane; j < NPAD_; j += 32) mx = fmaxf(mx, lds[r][j] + biasp[j]);
    for (int s = 16; s; s >>= 1) mx = fmaxf(mx, __shfl_xor(mx, s, 32));
    float se = 0.f;
    for (int j = lane; j < NPAD_; j += 32) se += expf((lds[r][j] + biasp[j]) - mx);
    for (int s = 16; s; s >>= 1) se += __shfl_xor(se, s, 32);
    if (lane == 0) {
      int tg = tgt[rowBase + r];
      if (tg >= 0) {
        float l = lds[r][tg] + biasp[tg];
        atomicAdd(nll_acc, mx + logf(se) - l);
      }
    }
  }
}

// -------- final scalar combine --------
__global__ void k_final(const float* acc, const int* cnt,
                        const float* s_det, const float* s_id, float* out) {
  float pos = acc[0], neg = acc[1], np = acc[2];
  float hm_loss = (np > 0.f) ? (-(pos + neg) / fmaxf(np, 1.f)) : (-neg);
  float denom = acc[5] * 2.f + 1e-4f;
  float wh_loss  = acc[3] / denom;
  float off_loss = acc[4] / denom;
  float reid = 0.f;
  for (int c = 0; c < 2; ++c) {
    int cn = cnt[c];
    if (cn > 0) {
      float nvalid = (float)min(cn, KG_);
      float ce = acc[6 + c] / fmaxf(nvalid, 1.f);
      reid += ce / fmaxf((float)cn, 1.f);
    }
  }
  float det = 1.0f * hm_loss + 0.1f * wh_loss + 1.0f * off_loss;
  float sd = s_det[0], si = s_id[0];
  float loss = 0.5f * (expf(-sd) * det + expf(-si) * reid + sd + si);
  out[0] = loss; out[1] = hm_loss; out[2] = wh_loss; out[3] = off_loss; out[4] = reid;
}

extern "C" void kernel_launch(void* const* d_in, const int* in_sizes, int n_in,
                              void* d_out, int out_size, void* d_ws, size_t ws_size,
                              hipStream_t stream) {
  const float* hm_pred   = (const float*)d_in[0];
  const float* wh_pred   = (const float*)d_in[1];
  const float* reg_pred  = (const float*)d_in[2];
  const float* id_pred   = (const float*)d_in[3];
  const float* hm        = (const float*)d_in[4];
  const float* wh        = (const float*)d_in[5];
  const float* reg       = (const float*)d_in[6];
  const float* reg_mask  = (const float*)d_in[7];
  const int*   ind       = (const int*)d_in[8];
  const int*   cls_id_map= (const int*)d_in[9];
  const int*   cls_tr_ids= (const int*)d_in[10];
  const float* W0        = (const float*)d_in[11];
  const float* b0        = (const float*)d_in[12];
  const float* W1        = (const float*)d_in[13];
  const float* b1        = (const float*)d_in[14];
  const float* s_det     = (const float*)d_in[15];
  const float* s_id      = (const float*)d_in[16];

  char* ws = (char*)d_ws;
  float*    acc    = (float*)(ws + OFF_ACC);
  int*      cnt    = ((int*)(ws + OFF_ACC)) + 8;
  int*      idxbuf = (int*)(ws + OFF_IDX);
  int*      tgtbuf = (int*)(ws + OFF_TGT);
  _Float16* W0h    = (_Float16*)(ws + OFF_W0H);
  _Float16* W1h    = (_Float16*)(ws + OFF_W1H);
  float*    biasp  = (float*)(ws + OFF_BP);
  _Float16* emb    = (_Float16*)(ws + OFF_EMB);

  k_init<<<4096, 256, 0, stream>>>(acc, tgtbuf, emb);
  k_convw<<<(2 * NPAD_ * EMB_ + 255) / 256, 256, 0, stream>>>(W0, b0, W1, b1, W0h, W1h, biasp);
  k_focal<<<(B_ * HM_C_ * HW_ + 255) / 256, 256, 0, stream>>>(hm_pred, hm, acc);
  k_regl1<<<1, 1024, 0, stream>>>(wh_pred, reg_pred, wh, reg, reg_mask, ind, acc);
  k_compact<<<(B_ * HW_ + 255) / 256, 256, 0, stream>>>(cls_id_map, cls_tr_ids, cnt, idxbuf, tgtbuf);
  dim3 gg(KG_, 2);
  k_gathernorm<<<gg, 128, 0, stream>>>(id_pred, cnt, idxbuf, emb);
  k_gemm<<<KG_ / 16, 32, 0, stream>>>(emb, W0h, biasp, tgtbuf, cnt + 0, acc + 6);
  k_gemm<<<KG_ / 16, 32, 0, stream>>>(emb + (size_t)KG_ * EMB_, W1h, biasp + NPAD_,
                                      tgtbuf + KG_, cnt + 1, acc + 7);
  k_final<<<1, 1, 0, stream>>>(acc, cnt, s_det, s_id, (float*)d_out);
}
